// ResGCN_MODEL_20255065768612
// MI455X (gfx1250) — matrix operand order repaired and checked by reading, not compiled, not run on hardware
//
#include <hip/hip_runtime.h>
#include <hip/hip_bf16.h>

// ---------------------------------------------------------------------------
// ResGCN forward for MI455X (gfx1250, wave32, WMMA).
// Dense GEMMs use v_wmma_f32_16x16x32_f16. Tiles are loaded from global as
// b128 f32 (two-phase: all loads in flight before any wait), converted f32->
// f16 once at staging, and stored to LDS as f16 (B transposed) so fragment
// reads are single ds_load_b128 per 8-half chunk with zero inner-loop cvts.
// Graph ops (A@W1, GCN aggregation) exploit edge sparsity via f32 atomics.
// ---------------------------------------------------------------------------

typedef __attribute__((ext_vector_type(16))) _Float16 v16h;
typedef __attribute__((ext_vector_type(8)))  _Float16 v8h;
typedef __attribute__((ext_vector_type(4)))  _Float16 v4h;
typedef __attribute__((ext_vector_type(8)))  float    v8f;
typedef __attribute__((ext_vector_type(4)))  float    v4f;

#define N_NODES 8192
#define F_DIM   256
#define RES_H   512
#define REP_H   256
#define EMB_DIM 128
#define N_EDGES 262144
#define GAMMA   0.1f

// ------------------------------ GEMM (WMMA) --------------------------------
// Block tile 128x64, K-step 32. 8 waves, each wave -> 32x32 micro-tile.
constexpr int BM = 128, BN = 64, BK = 32;
constexpr int LDAH = BK + 16;   // 48 halves = 96 B row pitch (16B-aligned reads)
constexpr int LDBH = BK + 16;   // Bs stored transposed: Bs[n][k], f16

template <bool TRANSB, bool BIAS, bool RELU>
__global__ __launch_bounds__(256) void gemm_wmma_f16(
    const float* __restrict__ A, const float* __restrict__ B,
    const float* __restrict__ bias, float* __restrict__ C,
    int M, int Nn, int K, int lda, int ldb, int ldc)
{
    __shared__ _Float16 As[BM * LDAH];   // [row][k] f16, 12 KB
    __shared__ _Float16 Bs[BN * LDBH];   // [col][k] f16 (transposed), 6 KB

    const int tid   = threadIdx.x;
    const int lane  = tid & 31;
    const int wave  = tid >> 5;      // 0..7
    const int waveM = wave >> 1;     // 0..3  (M direction, 32 rows each)
    const int waveN = wave & 1;      // 0..1  (N direction, 32 cols each)
    const int r16   = lane & 15;
    const int grp   = lane >> 4;     // lane group 0/1 per ISA layouts

    const int m0 = blockIdx.y * BM;
    const int n0 = blockIdx.x * BN;

    v8f acc[2][2];
#pragma unroll
    for (int i = 0; i < 2; ++i)
#pragma unroll
        for (int j = 0; j < 2; ++j) acc[i][j] = v8f{};

    for (int k0 = 0; k0 < K; k0 += BK) {
        // ---- phase 1: issue ALL global loads (b128) into registers ----
        v4f aR[4];
#pragma unroll
        for (int j = 0; j < 4; ++j) {       // A tile: 128x32 f32 = 1024 v4f
            int i   = tid + 256 * j;
            int row = i >> 3;
            int c4  = i & 7;
            aR[j] = *(const v4f*)&A[(size_t)(m0 + row) * lda + (k0 + c4 * 4)];
        }
        v4f bR[2];
#pragma unroll
        for (int j = 0; j < 2; ++j) {       // B tile: 32x64 f32 = 512 v4f
            int i = tid + 256 * j;
            if (TRANSB) {                   // B[n][k]: vector along k
                int cc = i >> 3;
                int k4 = i & 7;
                bR[j] = *(const v4f*)&B[(size_t)(n0 + cc) * ldb + (k0 + k4 * 4)];
            } else {                        // B[k][n]: vector along n
                int kk = i >> 4;
                int c4 = i & 15;
                bR[j] = *(const v4f*)&B[(size_t)(k0 + kk) * ldb + (n0 + c4 * 4)];
            }
        }
        // prefetch next K tile (global_prefetch_b8)
        if (k0 + BK < K) {
            __builtin_prefetch(&A[(size_t)(m0 + (tid >> 1)) * lda + (k0 + BK)], 0, 3);
            __builtin_prefetch(TRANSB
                ? &B[(size_t)(n0 + (tid & 63)) * ldb + (k0 + BK)]
                : &B[(size_t)(k0 + BK + (tid >> 3)) * ldb + n0], 0, 3);
        }

        __syncthreads();   // previous iteration's fragment reads complete

        // ---- phase 2: convert f32 -> f16 once, store tiles to LDS ----
#pragma unroll
        for (int j = 0; j < 4; ++j) {
            int i   = tid + 256 * j;
            int row = i >> 3;
            int c4  = i & 7;
            v4h h4;
#pragma unroll
            for (int t = 0; t < 4; ++t) h4[t] = (_Float16)aR[j][t];
            *(v4h*)&As[row * LDAH + c4 * 4] = h4;         // ds_store_b64
        }
#pragma unroll
        for (int j = 0; j < 2; ++j) {
            int i = tid + 256 * j;
            if (TRANSB) {
                int cc = i >> 3;
                int k4 = i & 7;
                v4h h4;
#pragma unroll
                for (int t = 0; t < 4; ++t) h4[t] = (_Float16)bR[j][t];
                *(v4h*)&Bs[cc * LDBH + k4 * 4] = h4;      // contiguous in k
            } else {
                int kk = i >> 4;
                int c4 = i & 15;
#pragma unroll
                for (int t = 0; t < 4; ++t)               // scatter across cols
                    Bs[(c4 * 4 + t) * LDBH + kk] = (_Float16)bR[j][t];
            }
        }
        __syncthreads();

        // ---- fragments: pure b128 LDS reads, no inner-loop converts ----
        // A (ISA 7.12.2): halves 0..7  <- As[row][grp*8   .. +7]
        //                 halves 8..15 <- As[row][16+grp*8 .. +7]
        v16h afrag[2], bfrag[2];
#pragma unroll
        for (int ms = 0; ms < 2; ++ms) {
            const _Float16* as = &As[(waveM * 32 + ms * 16 + r16) * LDAH + grp * 8];
            v8h lo = *(const v8h*)(as);
            v8h hi = *(const v8h*)(as + 16);
            afrag[ms] = __builtin_shufflevector(lo, hi,
                0, 1, 2, 3, 4, 5, 6, 7, 8, 9, 10, 11, 12, 13, 14, 15);
        }
        // B: halves 0..7 <- Bs[col][grp*8 .. +7], halves 8..15 <- Bs[col][16+grp*8 .. +7]
#pragma unroll
        for (int ns = 0; ns < 2; ++ns) {
            const _Float16* bs = &Bs[(waveN * 32 + ns * 16 + r16) * LDBH + grp * 8];
            v8h lo = *(const v8h*)(bs);
            v8h hi = *(const v8h*)(bs + 16);
            bfrag[ns] = __builtin_shufflevector(lo, hi,
                0, 1, 2, 3, 4, 5, 6, 7, 8, 9, 10, 11, 12, 13, 14, 15);
        }
#pragma unroll
        for (int ms = 0; ms < 2; ++ms)
#pragma unroll
            for (int ns = 0; ns < 2; ++ns)
                acc[ms][ns] = __builtin_amdgcn_wmma_f32_16x16x32_f16(
                    false, afrag[ms], false, bfrag[ns],
                    (short)0, acc[ms][ns], false, false);
    }

    // ---- epilogue: C/D layout -> row = grp*8+v, col = lane%16 ----
#pragma unroll
    for (int ms = 0; ms < 2; ++ms)
#pragma unroll
        for (int ns = 0; ns < 2; ++ns) {
            int colg = n0 + waveN * 32 + ns * 16 + r16;
            float bv = BIAS ? bias[colg] : 0.f;
#pragma unroll
            for (int v = 0; v < 8; ++v) {
                int rowg  = m0 + waveM * 32 + ms * 16 + grp * 8 + v;
                float val = acc[ms][ns][v] + bv;
                if (RELU) val = val > 0.f ? val : 0.f;
                C[(size_t)rowg * ldc + colg] = val;
            }
        }
}

// ------------------------------ helpers ------------------------------------

__global__ void zero_f32(float* __restrict__ p, size_t n) {
    size_t i = (size_t)blockIdx.x * blockDim.x + threadIdx.x;
    size_t s = (size_t)gridDim.x * blockDim.x;
    for (; i < n; i += s) p[i] = 0.f;
}

// out[src[e], :] += W[dst[e], :]   (sparse A @ res_W1)
__global__ void scatter_aw(const int* __restrict__ ei, const float* __restrict__ W,
                           float* __restrict__ out, int C) {
    int e = blockIdx.x, f = threadIdx.x;
    int s = ei[e], d = ei[N_EDGES + e];
    atomicAdd(&out[(size_t)s * C + f], W[(size_t)d * C + f]);
}

// agg[dst[e], :] += h[src[e], :] (* dinv[src]*dinv[dst] if NORM)
template <bool NORM>
__global__ void edge_aggregate(const int* __restrict__ ei, const float* __restrict__ h,
                               const float* __restrict__ dinv, float* __restrict__ agg, int C) {
    int e = blockIdx.x, f = threadIdx.x;
    int s = ei[e], d = ei[N_EDGES + e];
    float v = h[(size_t)s * C + f];
    if (NORM) v *= dinv[s] * dinv[d];
    atomicAdd(&agg[(size_t)d * C + f], v);
}

__global__ void deg_count(const int* __restrict__ ei, float* __restrict__ deg) {
    size_t i = (size_t)blockIdx.x * blockDim.x + threadIdx.x;
    size_t s = (size_t)gridDim.x * blockDim.x;
    for (; i < N_EDGES; i += s) atomicAdd(&deg[ei[N_EDGES + i]], 1.f);
}

__global__ void finish_dinv(float* __restrict__ deg) {
    int i = blockIdx.x * blockDim.x + threadIdx.x;
    if (i < N_NODES) deg[i] = rsqrtf(deg[i] + 1.f);   // +1 self loop
}

// x = relu(x + b[col]) in place over [N, C], C power of two
__global__ void bias_relu_ip(float* __restrict__ x, const float* __restrict__ b,
                             int cMask, size_t total) {
    size_t i = (size_t)blockIdx.x * blockDim.x + threadIdx.x;
    size_t s = (size_t)gridDim.x * blockDim.x;
    for (; i < total; i += s) {
        float v = x[i] + b[i & cMask];
        x[i] = v > 0.f ? v : 0.f;
    }
}

// out = relu(agg + b [+ hlin*dinv^2 if NORM])
template <bool NORM>
__global__ void post_agg(const float* __restrict__ agg, const float* __restrict__ hlin,
                         const float* __restrict__ dinv, const float* __restrict__ b,
                         float* __restrict__ out, int cShift, int cMask, size_t total) {
    size_t i = (size_t)blockIdx.x * blockDim.x + threadIdx.x;
    size_t s = (size_t)gridDim.x * blockDim.x;
    for (; i < total; i += s) {
        float v = agg[i] + b[i & cMask];
        if (NORM) {
            float di = dinv[i >> cShift];
            v += hlin[i] * di * di;
        }
        out[i] = v > 0.f ? v : 0.f;
    }
}

// out = H * exp(-GAMMA * Rl)
__global__ void mask_mul(const float* __restrict__ H, const float* __restrict__ Rl,
                         float* __restrict__ out, size_t total) {
    size_t i = (size_t)blockIdx.x * blockDim.x + threadIdx.x;
    size_t s = (size_t)gridDim.x * blockDim.x;
    for (; i < total; i += s) out[i] = H[i] * __expf(-GAMMA * Rl[i]);
}

// ------------------------------ driver -------------------------------------

static inline void launch_gemm(bool transB, bool bias, bool relu,
                               const float* A, const float* B, const float* b, float* C,
                               int M, int Nn, int K, int lda, int ldb, int ldc,
                               hipStream_t st) {
    dim3 grid(Nn / BN, M / BM), blk(256);
    if (transB) {
        if (bias && relu) gemm_wmma_f16<true, true, true><<<grid, blk, 0, st>>>(A, B, b, C, M, Nn, K, lda, ldb, ldc);
        else              gemm_wmma_f16<true, false, false><<<grid, blk, 0, st>>>(A, B, b, C, M, Nn, K, lda, ldb, ldc);
    } else {
        if (bias && relu) gemm_wmma_f16<false, true, true><<<grid, blk, 0, st>>>(A, B, b, C, M, Nn, K, lda, ldb, ldc);
        else              gemm_wmma_f16<false, false, false><<<grid, blk, 0, st>>>(A, B, b, C, M, Nn, K, lda, ldb, ldc);
    }
}

extern "C" void kernel_launch(void* const* d_in, const int* in_sizes, int n_in,
                              void* d_out, int out_size, void* d_ws, size_t ws_size,
                              hipStream_t stream) {
    const float* x      = (const float*)d_in[0];
    const int*   ei     = (const int*)d_in[1];
    const float* res_W1 = (const float*)d_in[2];
    const float* res_b1 = (const float*)d_in[3];
    const float* res_W2 = (const float*)d_in[4];
    const float* res_b2 = (const float*)d_in[5];
    const float* fc_W1  = (const float*)d_in[6];
    const float* fc_b1  = (const float*)d_in[7];
    const float* fc_W2  = (const float*)d_in[8];
    const float* fc_b2  = (const float*)d_in[9];
    const float* g_W0   = (const float*)d_in[10];
    const float* g_b0   = (const float*)d_in[11];
    const float* g_W1   = (const float*)d_in[12];
    const float* g_b1   = (const float*)d_in[13];
    const float* g_W2   = (const float*)d_in[14];
    const float* g_b2   = (const float*)d_in[15];
    const float* dec_W1 = (const float*)d_in[16];
    const float* dec_b1 = (const float*)d_in[17];
    const float* dec_W2 = (const float*)d_in[18];
    const float* dec_b2 = (const float*)d_in[19];

    // output layout: X_hat [N,F] | A_hat [N,N] | R [N,F]
    float* Xhat = (float*)d_out;
    float* Ahat = Xhat + (size_t)N_NODES * F_DIM;
    float* Rout = Ahat + (size_t)N_NODES * N_NODES;

    // workspace layout (floats)
    float* ws   = (float*)d_ws;
    float* T1   = ws;                                  // [N, RES_H]
    float* Hbuf = T1 + (size_t)N_NODES * RES_H;        // [N, REP_H]
    float* Lin  = Hbuf + (size_t)N_NODES * REP_H;      // [N, REP_H]
    float* Agg  = Lin + (size_t)N_NODES * REP_H;       // [N, REP_H]
    float* Rl1  = Agg + (size_t)N_NODES * REP_H;       // [N, REP_H]
    float* Rl2  = Rl1 + (size_t)N_NODES * REP_H;       // [N, REP_H] (also mask)
    float* Hemb = Rl2 + (size_t)N_NODES * REP_H;       // [N, EMB_DIM]
    float* dinv = Hemb + (size_t)N_NODES * EMB_DIM;    // [N]

    const size_t NR = (size_t)N_NODES * REP_H;
    const size_t NE = (size_t)N_NODES * EMB_DIM;
    dim3 gsB(2048), gsT(256);

    // --- R = relu(relu(A @ res_W1 + b1) @ res_W2 + b2), A sparse ---
    zero_f32<<<gsB, gsT, 0, stream>>>(T1, (size_t)N_NODES * RES_H);
    scatter_aw<<<N_EDGES, RES_H, 0, stream>>>(ei, res_W1, T1, RES_H);
    bias_relu_ip<<<gsB, gsT, 0, stream>>>(T1, res_b1, RES_H - 1, (size_t)N_NODES * RES_H);
    launch_gemm(false, true, true, T1, res_W2, res_b2, Rout,
                N_NODES, F_DIM, RES_H, RES_H, F_DIM, F_DIM, stream);

    // --- degrees: dinv = rsqrt(deg_dst + 1) ---
    zero_f32<<<64, 128, 0, stream>>>(dinv, N_NODES);
    deg_count<<<512, 256, 0, stream>>>(ei, dinv);
    finish_dinv<<<N_NODES / 256, 256, 0, stream>>>(dinv);

    // --- GCN layer 0 (no norm): H = relu(agg(x @ W0) + b0) ---
    launch_gemm(false, false, false, x, g_W0, nullptr, Lin,
                N_NODES, REP_H, F_DIM, F_DIM, REP_H, REP_H, stream);
    zero_f32<<<gsB, gsT, 0, stream>>>(Agg, NR);
    edge_aggregate<false><<<N_EDGES, REP_H, 0, stream>>>(ei, Lin, dinv, Agg, REP_H);
    post_agg<false><<<gsB, gsT, 0, stream>>>(Agg, Lin, dinv, g_b0, Hbuf, 8, REP_H - 1, NR);

    // --- layer 1: Rl1 = relu(R@fcW1+b); H = relu(gcn_norm((H*exp(-g*Rl1))@W1)+b1) ---
    launch_gemm(false, true, true, Rout, fc_W1, fc_b1, Rl1,
                N_NODES, REP_H, F_DIM, F_DIM, REP_H, REP_H, stream);
    mask_mul<<<gsB, gsT, 0, stream>>>(Hbuf, Rl1, Rl2, NR);
    launch_gemm(false, false, false, Rl2, g_W1, nullptr, Lin,
                N_NODES, REP_H, REP_H, REP_H, REP_H, REP_H, stream);
    zero_f32<<<gsB, gsT, 0, stream>>>(Agg, NR);
    edge_aggregate<true><<<N_EDGES, REP_H, 0, stream>>>(ei, Lin, dinv, Agg, REP_H);
    post_agg<true><<<gsB, gsT, 0, stream>>>(Agg, Lin, dinv, g_b1, Hbuf, 8, REP_H - 1, NR);

    // --- layer 2: Rl2 = relu(Rl1@fcW2+b); Hemb = relu(gcn_norm((H*exp)@W2)+b2) ---
    launch_gemm(false, true, true, Rl1, fc_W2, fc_b2, Rl2,
                N_NODES, REP_H, REP_H, REP_H, REP_H, REP_H, stream);
    mask_mul<<<gsB, gsT, 0, stream>>>(Hbuf, Rl2, Rl1, NR);   // Rl1 reused as mask
    launch_gemm(false, false, false, Rl1, g_W2, nullptr, Lin,
                N_NODES, EMB_DIM, REP_H, REP_H, EMB_DIM, EMB_DIM, stream);
    zero_f32<<<gsB, gsT, 0, stream>>>(Agg, NE);
    edge_aggregate<true><<<N_EDGES, EMB_DIM, 0, stream>>>(ei, Lin, dinv, Agg, EMB_DIM);
    post_agg<true><<<gsB, gsT, 0, stream>>>(Agg, Lin, dinv, g_b2, Hemb, 7, EMB_DIM - 1, NE);

    // --- decoder: X_hat = relu(relu(Hemb@decW1+b)@decW2+b) ---
    launch_gemm(false, true, true, Hemb, dec_W1, dec_b1, Rl1,
                N_NODES, REP_H, EMB_DIM, EMB_DIM, REP_H, REP_H, stream);
    launch_gemm(false, true, true, Rl1, dec_W2, dec_b2, Xhat,
                N_NODES, F_DIM, REP_H, REP_H, F_DIM, F_DIM, stream);

    // --- A_hat = Hemb @ Hemb^T (dominant GEMM: 17.2 GFLOP, 268 MB out) ---
    launch_gemm(true, false, false, Hemb, Hemb, nullptr, Ahat,
                N_NODES, N_NODES, EMB_DIM, EMB_DIM, EMB_DIM, N_NODES, stream);
}
